// Attention_85169201480311
// MI455X (gfx1250) — matrix operand order repaired and checked by reading, not compiled, not run on hardware
//
#include <hip/hip_runtime.h>

typedef __attribute__((ext_vector_type(16))) _Float16 v16h;
typedef __attribute__((ext_vector_type(8)))  _Float16 v8h;
typedef __attribute__((ext_vector_type(16))) __bf16   v16bf;
typedef __attribute__((ext_vector_type(8)))  __bf16   v8bf;
typedef __attribute__((ext_vector_type(8)))  float    v8f;
typedef __attribute__((ext_vector_type(4)))  unsigned u32x4;
typedef __attribute__((ext_vector_type(8)))  unsigned u32x8;

#define NPIX  1024
#define CCH   256
#define HEADS 8
#define DHD   32
#define BATCH 8
#define KDIM  2304                      /* C * 9 */
#define ATT_SCALE 0.17677669529663687f  /* 32^-0.5 */

static __device__ __forceinline__ __bf16 f2bf(float f) {
  union { float f; unsigned u; } c; c.f = f;
  unsigned r = (c.u + 0x7FFFu + ((c.u >> 16) & 1u)) >> 16;
  union { unsigned short s; __bf16 b; } o;
  o.s = (unsigned short)r;
  return o.b;
}

// ---------------------------------------------------------------------------
// Tensor Data Mover helpers (CDNA5 ISA ch.7/8: D# groups in SGPRs).
// group0: [1:0]=count=1, [63:32]=lds_addr, [120:64]=global_addr, [127:126]=2
// group1: [17:16]=data_size(1->2B), [79:48]=tensor_dim0, [111:80]=tensor_dim1,
//         [127:112]=tile_dim0, [143:128]=tile_dim1, [207:160]=dim0_stride
// ---------------------------------------------------------------------------
static __device__ __forceinline__ u32x4 tdm_g0(unsigned lds_addr,
                                               const void* gaddr) {
  unsigned long long ga = (unsigned long long)gaddr;
  u32x4 g;
  g.x = 1u;                                            // count = 1 (valid D#)
  g.y = lds_addr;                                      // LDS byte address
  g.z = (unsigned)ga;                                  // global_addr[31:0]
  g.w = (unsigned)((ga >> 32) & 0x1FFFFFFu) | (2u << 30);  // addr[56:32]|type=2
  return g;
}
static __device__ __forceinline__ u32x8 tdm_g1(unsigned dim0, unsigned dim1,
                                               unsigned t0, unsigned t1,
                                               unsigned long long s0) {
  u32x8 g;
  g[0] = 0x10000u;                                     // data_size = 1 (2 B)
  g[1] = (dim0 & 0xFFFFu) << 16;                       // tensor_dim0[15:0]@48
  g[2] = ((dim0 >> 16) & 0xFFFFu) | ((dim1 & 0xFFFFu) << 16);
  g[3] = ((dim1 >> 16) & 0xFFFFu) | ((t0 & 0xFFFFu) << 16);  // tile_dim0@112
  g[4] = (t1 & 0xFFFFu);                               // tile_dim1 (tile2=0)
  g[5] = (unsigned)(s0 & 0xFFFFFFFFull);               // dim0_stride[31:0]
  g[6] = (unsigned)((s0 >> 32) & 0xFFFFull);           // dim0_stride[47:32]
  g[7] = 0u;                                           // dim1_stride unused
  return g;
}
static __device__ __forceinline__ void tdm_load(u32x4 g0, u32x8 g1) {
  asm volatile("tensor_load_to_lds %0, %1" :: "s"(g0), "s"(g1) : "memory");
}

// ---------------------------------------------------------------------------
// im2col: pat[b][pix][k] (bf16), k = cin*9 + kh*3 + kw. Paid once, shared by
// all three conv GEMMs; removes all divergent gather math from inner loops.
// ---------------------------------------------------------------------------
__global__ __launch_bounds__(256) void im2col_kernel(
    const float* __restrict__ x, __bf16* __restrict__ pat) {
  size_t idx = (size_t)blockIdx.x * 256 + threadIdx.x;
  int k = (int)(idx % KDIM);
  size_t rest = idx / KDIM;
  int pix = (int)(rest & 1023);
  int b = (int)(rest >> 10);
  int cin = k / 9;
  int pos = k - cin * 9;
  int kh = pos / 3, kw = pos - kh * 3;
  int yy = (pix >> 5) + kh - 1, xx = (pix & 31) + kw - 1;
  float v = 0.0f;
  if ((unsigned)yy < 32u && (unsigned)xx < 32u)
    v = x[((size_t)b * CCH + cin) * NPIX + yy * 32 + xx];
  pat[idx] = f2bf(v);
}

__global__ __launch_bounds__(256) void cast_kernel(
    const float* __restrict__ s, __bf16* __restrict__ d, int n) {
  int i = blockIdx.x * 256 + threadIdx.x;
  if (i < n) d[i] = f2bf(s[i]);
}

// ---------------------------------------------------------------------------
// GEMM wave tile: 32 rows(M=pix) x 32 cols(N=cout), K in steps of 32.
// 4 WMMAs per K-step on 8 b128 loads; next step's fragments are loaded
// before this step's WMMAs issue (register double-buffer) so the matrix
// pipe overlaps load latency instead of s_wait_loadcnt 0 per WMMA.
// Used for both the 3x3 convs (kd=2304) and the 1x1 out-proj (kd=256).
// ---------------------------------------------------------------------------
__global__ __launch_bounds__(32) void gemm_kernel(
    const __bf16* __restrict__ amat, const __bf16* __restrict__ bmat,
    const float* __restrict__ bias, float* __restrict__ y, int kd) {
  int b = blockIdx.x, ct = blockIdx.y, pt = blockIdx.z;
  int lane = threadIdx.x, ln = lane & 15, hi = lane >> 4;
  int coutBase = ct * 32, pixBase = pt * 32;
  const __bf16* arow0 = amat + ((size_t)b * NPIX + pixBase + ln) * kd;
  const __bf16* arow1 = arow0 + (size_t)16 * kd;
  const __bf16* brow0 = bmat + (size_t)(coutBase + ln) * kd;
  const __bf16* brow1 = brow0 + (size_t)16 * kd;

  auto loadA = [&](const __bf16* row, int kb) {
    v16bf a = *(const v16bf*)(row + kb + hi * 16);      // A: K = kb+j+16*hi
    return a;
  };
  auto loadB = [&](const __bf16* row, int kb) {
    return *(const v16bf*)(row + kb + hi * 16);         // B: K = kb+j+16*hi
  };

  v8f acc00 = {}, acc01 = {}, acc10 = {}, acc11 = {};
  v16bf a0 = loadA(arow0, 0), a1 = loadA(arow1, 0);
  v16bf b0 = loadB(brow0, 0), b1 = loadB(brow1, 0);

  for (int kb = 0; kb < kd; kb += 32) {
    int nk = (kb + 32 < kd) ? kb + 32 : 0;  // last iter reloads k=0 (unused)
    if (kb + 64 < kd) __builtin_prefetch(arow0 + kb + 64, 0, 3);
    v16bf na0 = loadA(arow0, nk);
    v16bf na1 = loadA(arow1, nk);
    v16bf nb0 = loadB(brow0, nk);
    v16bf nb1 = loadB(brow1, nk);
    acc00 = __builtin_amdgcn_wmma_f32_16x16x32_bf16(false, a0, false, b0,
                                                    (short)0, acc00, false, false);
    acc01 = __builtin_amdgcn_wmma_f32_16x16x32_bf16(false, a0, false, b1,
                                                    (short)0, acc01, false, false);
    acc10 = __builtin_amdgcn_wmma_f32_16x16x32_bf16(false, a1, false, b0,
                                                    (short)0, acc10, false, false);
    acc11 = __builtin_amdgcn_wmma_f32_16x16x32_bf16(false, a1, false, b1,
                                                    (short)0, acc11, false, false);
    a0 = na0; a1 = na1; b0 = nb0; b1 = nb1;
  }

  float bi0 = bias ? bias[coutBase + ln] : 0.0f;
  float bi1 = bias ? bias[coutBase + 16 + ln] : 0.0f;
  // C layout: col(cout)=lane%16, rows(pix)=r+8*hi
  float* y0 = y + ((size_t)b * CCH + coutBase + ln) * NPIX + pixBase + 8 * hi;
  float* y1 = y0 + (size_t)16 * NPIX;
#pragma unroll
  for (int r = 0; r < 8; ++r) {
    y0[r]      = acc00[r] + bi0;   // rows pixBase..  , cols coutBase..
    y0[r + 16] = acc10[r] + bi0;   // rows pixBase+16.
    y1[r]      = acc01[r] + bi1;   // cols coutBase+16.
    y1[r + 16] = acc11[r] + bi1;
  }
}

// ---------------------------------------------------------------------------
// GroupNorm stats (per-sample sum / sumsq) and normalize+GELU epilogue.
// ---------------------------------------------------------------------------
__global__ void zero_stats_kernel(float* stats) {
  if (threadIdx.x < 48) stats[threadIdx.x] = 0.0f;
}

__global__ __launch_bounds__(256) void stats_kernel(
    const float* __restrict__ y, float* __restrict__ st) {
  __shared__ float s1[256], s2[256];
  int b = blockIdx.y;
  size_t i = (size_t)b * CCH * NPIX + (size_t)blockIdx.x * 256 + threadIdx.x;
  float v = y[i];
  s1[threadIdx.x] = v;
  s2[threadIdx.x] = v * v;
  __syncthreads();
  for (int s = 128; s > 0; s >>= 1) {
    if ((int)threadIdx.x < s) {
      s1[threadIdx.x] += s1[threadIdx.x + s];
      s2[threadIdx.x] += s2[threadIdx.x + s];
    }
    __syncthreads();
  }
  if (threadIdx.x == 0) {
    atomicAdd(&st[b], s1[0]);
    atomicAdd(&st[8 + b], s2[0]);
  }
}

__global__ __launch_bounds__(256) void gn_gelu_kernel(
    const float* __restrict__ y, const float* __restrict__ g,
    const float* __restrict__ beta, const float* __restrict__ st,
    _Float16* __restrict__ out, int vlayout) {
  size_t i = (size_t)blockIdx.x * 256 + threadIdx.x;
  int b = (int)(i >> 18);
  int c = (int)((i >> 10) & 255);
  int n = (int)(i & 1023);
  const float M = (float)(CCH * NPIX);
  float mean = st[b] / M;
  float var = st[8 + b] / M - mean * mean;
  float v = y[i];
  v = (v - mean) * rsqrtf(var + 1e-6f) * g[c] + beta[c];
  v = 0.5f * v * (1.0f + erff(v * 0.70710678118654752f));
  int h = c >> 5, d = c & 31;
  size_t o = vlayout ? ((((size_t)b * HEADS + h) * DHD + d) * NPIX + n)
                     : ((((size_t)b * HEADS + h) * NPIX + n) * DHD + d);
  out[o] = (_Float16)v;
}

// ---------------------------------------------------------------------------
// Flash attention, f16 WMMA. 4 waves x 16 query rows. K/V^T 32-key chunks are
// double-buffered into LDS by the Tensor Data Mover (issued by wave 0,
// s_wait_tensorcnt + workgroup barrier hand-off), overlapping DMA with WMMA.
// ---------------------------------------------------------------------------
__global__ __launch_bounds__(128) void attn_kernel(
    const _Float16* __restrict__ qf, const _Float16* __restrict__ kf,
    const _Float16* __restrict__ vf, const float* __restrict__ bias_table,
    __bf16* __restrict__ ao) {
  int bh = blockIdx.x;            // b*8 + h
  int b = bh >> 3, h = bh & 7;
  int tid = threadIdx.x;
  int w = tid >> 5;
  int lane = tid & 31;
  int ln = lane & 15, hi = lane >> 4;
  int rowBase = blockIdx.y * 64 + w * 16;

  const _Float16* Q = qf + (size_t)bh * (NPIX * DHD);
  const _Float16* K = kf + (size_t)bh * (NPIX * DHD);
  const _Float16* V = vf + (size_t)bh * (DHD * NPIX);

  __shared__ _Float16 kbuf[2][32 * 32];   // [key_local][dh]
  __shared__ _Float16 vbuf[2][32 * 32];   // [dh][key_local]
  __shared__ _Float16 pl[4][16 * 32];     // per-wave P transpose buffer

  // Q A-fragment: M = lane%16, K(j) = j + 16*hi
  v16h aq = *(const v16h*)(Q + (size_t)(rowBase + ln) * DHD + hi * 16);

  float m_i[8], l_i[8];
  v8f o0 = {}, o1 = {};
#pragma unroll
  for (int r = 0; r < 8; ++r) { m_i[r] = -1e30f; l_i[r] = 0.0f; }

  int y1r[8], x1r[8];
#pragma unroll
  for (int r = 0; r < 8; ++r) {
    int nq = rowBase + r + 8 * hi;
    y1r[r] = nq >> 5; x1r[r] = nq & 31;
  }

  if (w == 0) {  // prime chunk 0 via TDM
    tdm_load(tdm_g0((unsigned)(size_t)&kbuf[0][0], K),
             tdm_g1(DHD, NPIX, 32, 32, DHD));
    tdm_load(tdm_g0((unsigned)(size_t)&vbuf[0][0], V),
             tdm_g1(NPIX, DHD, 32, 32, NPIX));
  }

  for (int kb = 0, it = 0; kb < NPIX; kb += 32, ++it) {
    int cur = it & 1;
    if (w == 0) {
      if (kb + 32 < NPIX) {  // launch next chunk, then wait for current
        tdm_load(tdm_g0((unsigned)(size_t)&kbuf[cur ^ 1][0],
                        K + (size_t)(kb + 32) * DHD),
                 tdm_g1(DHD, NPIX, 32, 32, DHD));
        tdm_load(tdm_g0((unsigned)(size_t)&vbuf[cur ^ 1][0], V + (kb + 32)),
                 tdm_g1(NPIX, DHD, 32, 32, NPIX));
        __builtin_amdgcn_s_wait_tensorcnt(2);  // in-order: chunk `it` landed
      } else {
        __builtin_amdgcn_s_wait_tensorcnt(0);
      }
    }
    __syncthreads();  // chunk `it` visible to all waves

    // K^T B-fragments from LDS: N = key(lane%16), K(dh) = j + 16*hi
    v16h bk0 = *(const v16h*)(&kbuf[cur][(size_t)ln * DHD + hi * 16]);
    v16h bk1 = *(const v16h*)(&kbuf[cur][(size_t)(16 + ln) * DHD + hi * 16]);
    v8f z = {};
    v8f s0 = __builtin_amdgcn_wmma_f32_16x16x32_f16(false, aq, false, bk0,
                                                    (short)0, z, false, false);
    v8f s1 = __builtin_amdgcn_wmma_f32_16x16x32_f16(false, aq, false, bk1,
                                                    (short)0, z, false, false);

    int col0 = kb + ln, col1 = col0 + 16;
    int y20 = col0 >> 5, x20 = col0 & 31;
    int y21 = col1 >> 5, x21 = col1 & 31;
    float p0[8], p1[8];
#pragma unroll
    for (int r = 0; r < 8; ++r) {
      float bb0 = bias_table[((y1r[r] - y20 + 31) * 63 + (x1r[r] - x20 + 31)) * HEADS + h];
      float bb1 = bias_table[((y1r[r] - y21 + 31) * 63 + (x1r[r] - x21 + 31)) * HEADS + h];
      float v0 = s0[r] * ATT_SCALE + bb0;
      float v1 = s1[r] * ATT_SCALE + bb1;
      float mx = fmaxf(v0, v1);
      mx = fmaxf(mx, __shfl_xor(mx, 1, 32));
      mx = fmaxf(mx, __shfl_xor(mx, 2, 32));
      mx = fmaxf(mx, __shfl_xor(mx, 4, 32));
      mx = fmaxf(mx, __shfl_xor(mx, 8, 32));
      float mnew = fmaxf(m_i[r], mx);
      float e0 = __expf(v0 - mnew);
      float e1 = __expf(v1 - mnew);
      float rs = e0 + e1;
      rs += __shfl_xor(rs, 1, 32);
      rs += __shfl_xor(rs, 2, 32);
      rs += __shfl_xor(rs, 4, 32);
      rs += __shfl_xor(rs, 8, 32);
      float corr = __expf(m_i[r] - mnew);
      l_i[r] = l_i[r] * corr + rs;
      m_i[r] = mnew;
      o0[r] *= corr;
      o1[r] *= corr;
      p0[r] = e0;
      p1[r] = e1;
    }
    // C-layout -> A-layout transpose through this wave's private LDS tile
#pragma unroll
    for (int r = 0; r < 8; ++r) {
      int row = r + 8 * hi;
      pl[w][row * 32 + ln]      = (_Float16)p0[r];
      pl[w][row * 32 + 16 + ln] = (_Float16)p1[r];
    }
    v16h ap;
    {
      const _Float16* base = &pl[w][ln * 32];
      int off = hi * 16;
#pragma unroll
      for (int j = 0; j < 16; ++j) {
        ap[j]     = base[j + off];
      }
    }
    // V B-fragments from LDS V^T chunk: N = dh(lane%16), K(key) = j + 16*hi
    v16h bv0 = *(const v16h*)(&vbuf[cur][(size_t)ln * 32 + hi * 16]);
    v16h bv1 = *(const v16h*)(&vbuf[cur][(size_t)(16 + ln) * 32 + hi * 16]);
    o0 = __builtin_amdgcn_wmma_f32_16x16x32_f16(false, ap, false, bv0,
                                                (short)0, o0, false, false);
    o1 = __builtin_amdgcn_wmma_f32_16x16x32_f16(false, ap, false, bv1,
                                                (short)0, o1, false, false);
    __syncthreads();  // all waves done with chunk `it` before TDM reuses buf
  }

  // write bf16 [B][N][C] (c = h*32+dh) for the 1x1 projection GEMM
  __bf16* ob = ao + (size_t)b * NPIX * CCH + h * DHD;
#pragma unroll
  for (int r = 0; r < 8; ++r) {
    float inv = 1.0f / l_i[r];
    int row = rowBase + r + 8 * hi;
    ob[(size_t)row * CCH + ln]      = f2bf(o0[r] * inv);
    ob[(size_t)row * CCH + 16 + ln] = f2bf(o1[r] * inv);
  }
}

// ---------------------------------------------------------------------------
extern "C" void kernel_launch(void* const* d_in, const int* in_sizes, int n_in,
                              void* d_out, int out_size, void* d_ws,
                              size_t ws_size, hipStream_t stream) {
  (void)in_sizes; (void)n_in; (void)out_size; (void)ws_size;
  const float* x          = (const float*)d_in[0];
  const float* Wq         = (const float*)d_in[1];
  const float* gq         = (const float*)d_in[2];
  const float* bq         = (const float*)d_in[3];
  const float* Wk         = (const float*)d_in[4];
  const float* gk         = (const float*)d_in[5];
  const float* bk         = (const float*)d_in[6];
  const float* Wv         = (const float*)d_in[7];
  const float* gv         = (const float*)d_in[8];
  const float* bv         = (const float*)d_in[9];
  const float* bias_table = (const float*)d_in[10];
  const float* Wout       = (const float*)d_in[11];
  const float* bout       = (const float*)d_in[12];
  float* out = (float*)d_out;

  char* ws = (char*)d_ws;
  size_t off = 0;
  __bf16* pat = (__bf16*)(ws + off); off += (size_t)BATCH * NPIX * KDIM * 2;
  __bf16* wbf = (__bf16*)(ws + off); off += (size_t)3 * CCH * KDIM * 2;
  __bf16* wobf = (__bf16*)(ws + off); off += (size_t)CCH * CCH * 2;
  float* y = (float*)(ws + off); off += (size_t)BATCH * CCH * NPIX * 4;
  _Float16* qf = (_Float16*)(ws + off); off += (size_t)BATCH * CCH * NPIX * 2;
  _Float16* kf = (_Float16*)(ws + off); off += (size_t)BATCH * CCH * NPIX * 2;
  _Float16* vf = (_Float16*)(ws + off); off += (size_t)BATCH * CCH * NPIX * 2;
  float* stats = (float*)(ws + off);
  __bf16* ao = (__bf16*)y;   // y is dead after the last gn_gelu; reuse (4MB<8MB)

  zero_stats_kernel<<<1, 64, 0, stream>>>(stats);
  im2col_kernel<<<(BATCH * NPIX * KDIM) / 256, 256, 0, stream>>>(x, pat);

  const int WCN = CCH * KDIM;            // 589824 per conv weight
  cast_kernel<<<(WCN + 255) / 256, 256, 0, stream>>>(Wq, wbf, WCN);
  cast_kernel<<<(WCN + 255) / 256, 256, 0, stream>>>(Wk, wbf + WCN, WCN);
  cast_kernel<<<(WCN + 255) / 256, 256, 0, stream>>>(Wv, wbf + 2 * WCN, WCN);
  cast_kernel<<<(CCH * CCH + 255) / 256, 256, 0, stream>>>(Wout, wobf,
                                                           CCH * CCH);

  const float* Gs[3] = {gq, gk, gv};
  const float* Bs[3] = {bq, bk, bv};
  _Float16*    Ts[3] = {qf, kf, vf};
  for (int p = 0; p < 3; ++p) {
    gemm_kernel<<<dim3(BATCH, 8, 32), 32, 0, stream>>>(
        pat, wbf + (size_t)p * WCN, nullptr, y, KDIM);
    stats_kernel<<<dim3(1024, BATCH), 256, 0, stream>>>(y, stats + p * 16);
    gn_gelu_kernel<<<(BATCH * CCH * NPIX) / 256, 256, 0, stream>>>(
        y, Gs[p], Bs[p], stats + p * 16, Ts[p], (p == 2) ? 1 : 0);
  }

  attn_kernel<<<dim3(64, 16), 128, 0, stream>>>(qf, kf, vf, bias_table, ao);

  // out-proj: bf16 ao[B][N][C] x Wout^T, reuse gemm_kernel with kd=256
  gemm_kernel<<<dim3(BATCH, 8, 32), 32, 0, stream>>>(ao, wobf, bout, out, CCH);
}
